// Attention_block_67473936220847
// MI455X (gfx1250) — matrix operand order repaired and checked by reading, not compiled, not run on hardware
//
#include <hip/hip_runtime.h>
#include <hip/hip_bf16.h>

// ---------------------------------------------------------------------------
// SAGAN-ish block stack for MI455X (gfx1250, wave32).
// Attention is flash-style: never materialize the [N,N] score matrix.
// Both attention GEMMs run on v_wmma_f32_16x16x32_bf16.
// Softmax row reductions use DPP16 butterflies (pure VALU, no LDS round-trip).
// ---------------------------------------------------------------------------

typedef __attribute__((ext_vector_type(16))) __bf16 v16bf;
typedef __attribute__((ext_vector_type(8)))  __bf16 v8bf;
typedef __attribute__((ext_vector_type(8)))  float  v8f;

#define WB   2      // batch
#define WN   4096   // H*W (64*64)
#define TPB  256    // 8 waves per workgroup
#define PSTL 72     // padded LDS row stride for P (bf16 elems) -> 144B, 16B-align ok

__device__ inline __bf16 f2bf(float f) {
  // round-to-nearest-even fp32 -> bf16 (used on the write-once projections)
  unsigned u = __builtin_bit_cast(unsigned, f);
  unsigned r = (u + 0x7FFFu + ((u >> 16) & 1u)) >> 16;
  return __builtin_bit_cast(__bf16, (unsigned short)r);
}

// Truncating fp32 -> bf16 (hot path: softmax probs in [0,1], <=2^-9 one-sided
// error on top of bf16 quantization; selects a single ds_store_b16_d16_hi).
__device__ inline __bf16 f2bf_trunc(float f) {
  unsigned u = __builtin_bit_cast(unsigned, f);
  return __builtin_bit_cast(__bf16, (unsigned short)(u >> 16));
}

__device__ inline void lds_fence() {
  asm volatile("s_wait_dscnt 0" ::: "memory");
}

// ---- DPP16 butterfly reduction over a 16-lane row (both halves independent).
// Step order keeps reduced groups uniform: xor1, xor2 (quad_perm), then
// row_half_mirror (combines quads), then row_mirror (combines 8-groups).
template <int CTRL>
__device__ inline float dppmov(float x) {
  int r = __builtin_amdgcn_update_dpp(0, __builtin_bit_cast(int, x),
                                      CTRL, 0xF, 0xF, true);
  return __builtin_bit_cast(float, r);
}
__device__ inline float red_max16(float x) {
  x = fmaxf(x, dppmov<0xB1>(x));   // quad_perm [1,0,3,2]  (xor 1)
  x = fmaxf(x, dppmov<0x4E>(x));   // quad_perm [2,3,0,1]  (xor 2)
  x = fmaxf(x, dppmov<0x141>(x));  // row_half_mirror      (combine quads)
  x = fmaxf(x, dppmov<0x140>(x));  // row_mirror           (combine 8-groups)
  return x;
}
__device__ inline float red_sum16(float x) {
  x += dppmov<0xB1>(x);
  x += dppmov<0x4E>(x);
  x += dppmov<0x141>(x);
  x += dppmov<0x140>(x);
  return x;
}

__device__ inline float block_reduce_add(float v, float* sh) {
  int tid = threadIdx.x;
  sh[tid] = v; __syncthreads();
  for (int s = TPB / 2; s > 0; s >>= 1) {
    if (tid < s) sh[tid] += sh[tid + s];
    __syncthreads();
  }
  float r = sh[0];
  __syncthreads();
  return r;
}

// ---------------------------------------------------------------------------
// conv1x1: y[b,co,n] = sum_ci x[b,ci,n] * w[ci,co] + bias[co]
// ---------------------------------------------------------------------------
__global__ __launch_bounds__(TPB)
void conv1x1_kernel(const float* __restrict__ x, const float* __restrict__ w,
                    const float* __restrict__ bias, float* __restrict__ y,
                    int Ci, int Co) {
  int i = blockIdx.x * TPB + threadIdx.x;
  int total = WB * Co * WN;
  if (i >= total) return;
  int n  = i & (WN - 1);
  int co = (i >> 12) % Co;
  int b  = i / (Co << 12);
  float a = bias[co];
  const float* xp = x + (size_t)b * Ci * WN + n;
  for (int ci = 0; ci < Ci; ++ci) a += xp[(size_t)ci * WN] * w[ci * Co + co];
  y[i] = a;
}

// ---------------------------------------------------------------------------
// BN batch stats (biased var over B*N) per channel
// ---------------------------------------------------------------------------
__global__ __launch_bounds__(TPB)
void bn_stats_kernel(const float* __restrict__ y, float* __restrict__ mean,
                     float* __restrict__ rstd, int C) {
  __shared__ float sh[TPB];
  int c = blockIdx.x;
  float s = 0.f, s2 = 0.f;
  for (int b = 0; b < WB; ++b) {
    const float* p = y + ((size_t)(b * C + c)) * WN;
    for (int i = threadIdx.x; i < WN; i += TPB) {
      float v = p[i]; s += v; s2 += v * v;
    }
  }
  s  = block_reduce_add(s, sh);
  s2 = block_reduce_add(s2, sh);
  if (threadIdx.x == 0) {
    float inv = 1.f / (float)(WB * WN);
    float m = s * inv;
    float var = s2 * inv - m * m;
    mean[c] = m;
    rstd[c] = rsqrtf(var + 1e-5f);
  }
}

// ---------------------------------------------------------------------------
// xn = relu((y-mean)*rstd*g + b)
// ---------------------------------------------------------------------------
__global__ __launch_bounds__(TPB)
void bn_relu_kernel(const float* __restrict__ y, const float* __restrict__ mean,
                    const float* __restrict__ rstd, const float* __restrict__ g,
                    const float* __restrict__ bb, float* __restrict__ xn, int C) {
  int i = blockIdx.x * TPB + threadIdx.x;
  int total = WB * C * WN;
  if (i >= total) return;
  int c = (i >> 12) % C;
  float v = (y[i] - mean[c]) * rstd[c] * g[c] + bb[c];
  xn[i] = fmaxf(v, 0.f);
}

// ---------------------------------------------------------------------------
// f/g projections -> transposed bf16 [B][N][32] rows, zero-padded cf -> 32.
// ---------------------------------------------------------------------------
template <int CF>
__global__ __launch_bounds__(TPB)
void proj_fg_kernel(const float* __restrict__ xn,
                    const float* __restrict__ wf, const float* __restrict__ bf,
                    const float* __restrict__ wg, const float* __restrict__ bg,
                    __bf16* __restrict__ fT, __bf16* __restrict__ gT, int C) {
  int idx = blockIdx.x * TPB + threadIdx.x;   // idx = b*WN + n, grid exact
  int n = idx & (WN - 1);
  int b = idx >> 12;
  float af[CF], ag[CF];
#pragma unroll
  for (int j = 0; j < CF; ++j) { af[j] = bf[j]; ag[j] = bg[j]; }
  const float* xp = xn + (size_t)b * C * WN + n;
  for (int c = 0; c < C; ++c) {
    float v = xp[(size_t)c * WN];
#pragma unroll
    for (int j = 0; j < CF; ++j) {
      af[j] += v * wf[c * CF + j];
      ag[j] += v * wg[c * CF + j];
    }
  }
  __bf16* fr = fT + ((size_t)idx << 5);
  __bf16* gr = gT + ((size_t)idx << 5);
#pragma unroll
  for (int j = 0; j < CF; ++j) { fr[j] = f2bf(af[j]); gr[j] = f2bf(ag[j]); }
#pragma unroll
  for (int j = CF; j < 32; ++j) { fr[j] = f2bf(0.f); gr[j] = f2bf(0.f); }
}

// ---------------------------------------------------------------------------
// h projection, bf16 channel-major [B][C][N]
// ---------------------------------------------------------------------------
__global__ __launch_bounds__(TPB)
void proj_h_kernel(const float* __restrict__ xn, const float* __restrict__ wh,
                   const float* __restrict__ bh, __bf16* __restrict__ hbf, int C) {
  int i = blockIdx.x * TPB + threadIdx.x;
  int total = WB * C * WN;
  if (i >= total) return;
  int n  = i & (WN - 1);
  int co = (i >> 12) % C;
  int b  = i / (C << 12);
  float a = bh[co];
  const float* xp = xn + (size_t)b * C * WN + n;
  for (int ci = 0; ci < C; ++ci) a += xp[(size_t)ci * WN] * wh[ci * C + co];
  hbf[i] = f2bf(a);
}

// ---------------------------------------------------------------------------
// Flash attention. One wave per 16-query tile; key loop in chunks of 64.
// Per chunk: 4 score tiles (WMMA), one DPP-butterfly softmax update over all
// 64 columns, P -> LDS (C/D layout -> A layout), then 2*CT accumulate WMMAs.
// All key-loop global addresses are single running pointers + immediate
// offsets so the backend strength-reduces the address math.
// out = gam * (o / l) + xn      (residual)
// ---------------------------------------------------------------------------
template <int CT>   // number of 16-channel tiles, C = 16*CT
__global__ __launch_bounds__(TPB)
void flash_attn_kernel(const __bf16* __restrict__ fT, const __bf16* __restrict__ gT,
                       const __bf16* __restrict__ hbf, const float* __restrict__ xn,
                       const float* __restrict__ gam, float* __restrict__ out) {
  const int C = CT * 16;
  __shared__ __bf16 ldsP[8][16 * PSTL];

  const int wave = threadIdx.x >> 5;
  const int lane = threadIdx.x & 31;
  const int l    = lane & 15;
  const int hi   = lane >> 4;
  const int qt   = blockIdx.x * 8 + wave;     // global 16-query tile id
  const int tilesPerB = WN >> 4;
  const int b  = qt / tilesPerB;
  const int n0 = (qt - b * tilesPerB) << 4;

  // A-fragment K offsets, contiguous order (matches B fills):
  // lanes 0-15: K 0-7 (v0-3) and 8-15 (v4-7); lanes 16-31: K 16-23 and 24-31
  const int kb0 = hi ? 16 : 0;
  const int kb1 = hi ? 24 : 8;

  // ---- load query A fragment from fT [b][n][32] bf16 ----
  const __bf16* frow = fT + ((size_t)(b * WN + n0 + l) << 5);
  v8bf a0 = *(const v8bf*)(frow + kb0);
  v8bf a1 = *(const v8bf*)(frow + kb1);
  v16bf afrag = __builtin_shufflevector(a0, a1, 0, 1, 2, 3, 4, 5, 6, 7,
                                                8, 9, 10, 11, 12, 13, 14, 15);

  float mrun[8], lrun[8];
  v8f acc[CT];
#pragma unroll
  for (int r = 0; r < 8; ++r) { mrun[r] = -3.0e38f; lrun[r] = 0.f; }
#pragma unroll
  for (int t = 0; t < CT; ++t) {
    v8f z = {0.f, 0.f, 0.f, 0.f, 0.f, 0.f, 0.f, 0.f};
    acc[t] = z;
  }

  // Running base pointers: subtile/half offsets become load immediates.
  const __bf16* gbase = gT + ((size_t)(b * WN + l) << 5) + hi * 16;
  const __bf16* hbase[CT];
#pragma unroll
  for (int t = 0; t < CT; ++t)
    hbase[t] = hbf + ((size_t)(b * C + t * 16 + l)) * WN + hi * 16;

  __bf16* pstore = &ldsP[wave][0];
  const __bf16* pload = &ldsP[wave][l * PSTL];

  for (int m0 = 0; m0 < WN; m0 += 64) {
    // ---- four 16x16 score tiles (64 keys) ----
    const __bf16* gm = gbase + (size_t)m0 * 32;   // +4096B per iteration
    v8f s[4];
#pragma unroll
    for (int sub = 0; sub < 4; ++sub) {
      v16bf bfrag = *(const v16bf*)(gm + sub * 512);   // imm offset sub*1024B
      v8f z = {0.f, 0.f, 0.f, 0.f, 0.f, 0.f, 0.f, 0.f};
      s[sub] = __builtin_amdgcn_wmma_f32_16x16x32_bf16(
          false, afrag, false, bfrag, (short)0, z, false, false);
    }

    // ---- online softmax across these 64 columns (DPP butterflies) ----
#pragma unroll
    for (int r = 0; r < 8; ++r) {
      float mx = fmaxf(fmaxf(s[0][r], s[1][r]), fmaxf(s[2][r], s[3][r]));
      mx = red_max16(mx);
      float mnew  = fmaxf(mrun[r], mx);
      float scale = __expf(mrun[r] - mnew);
      float p0 = __expf(s[0][r] - mnew);
      float p1 = __expf(s[1][r] - mnew);
      float p2 = __expf(s[2][r] - mnew);
      float p3 = __expf(s[3][r] - mnew);
      float rs = red_sum16((p0 + p1) + (p2 + p3));
      lrun[r] = lrun[r] * scale + rs;
      mrun[r] = mnew;
#pragma unroll
      for (int t = 0; t < CT; ++t) acc[t][r] *= scale;
      int row = r + 8 * hi;                    // C/D layout: row = r + 8*hi
      __bf16* pr = pstore + row * PSTL + l;    // col = sub*16 + l
      pr[0]  = f2bf_trunc(p0);
      pr[16] = f2bf_trunc(p1);
      pr[32] = f2bf_trunc(p2);
      pr[48] = f2bf_trunc(p3);
    }
    lds_fence();

    // ---- P as two A fragments (K 0..31 and 32..63), o += P x h ----
    v16bf pfrag[2];
#pragma unroll
    for (int half = 0; half < 2; ++half) {
      v8bf q0 = *(const v8bf*)(pload + half * 32 + kb0);
      v8bf q1 = *(const v8bf*)(pload + half * 32 + kb1);
      pfrag[half] = __builtin_shufflevector(q0, q1, 0, 1, 2, 3, 4, 5, 6, 7,
                                                    8, 9, 10, 11, 12, 13, 14, 15);
    }
#pragma unroll
    for (int t = 0; t < CT; ++t) {
      const __bf16* hm = hbase[t] + m0;        // +128B per iteration
      v16bf bh0 = *(const v16bf*)hm;           // keys m0   .. m0+31 (this half)
      v16bf bh1 = *(const v16bf*)(hm + 32);    // keys m0+32.. m0+63 (imm +64B)
      acc[t] = __builtin_amdgcn_wmma_f32_16x16x32_bf16(
          false, pfrag[0], false, bh0, (short)0, acc[t], false, false);
      acc[t] = __builtin_amdgcn_wmma_f32_16x16x32_bf16(
          false, pfrag[1], false, bh1, (short)0, acc[t], false, false);
    }
    lds_fence();
  }

  // ---- epilogue: out = gam * (o / l) + xn ----
  float ga = gam[0];
  float invl[8];
#pragma unroll
  for (int r = 0; r < 8; ++r) invl[r] = 1.0f / lrun[r];
#pragma unroll
  for (int t = 0; t < CT; ++t) {
    int c = t * 16 + l;
#pragma unroll
    for (int r = 0; r < 8; ++r) {
      int n = n0 + r + 8 * hi;
      size_t idx = ((size_t)(b * C + c)) * WN + n;
      out[idx] = ga * (acc[t][r] * invl[r]) + xn[idx];
    }
  }
}

// ---------------------------------------------------------------------------
// Final stage: conv(96->512, no bias) fused into stats and relu+GAP passes.
// ---------------------------------------------------------------------------
__global__ __launch_bounds__(TPB)
void final_stats_kernel(const float* __restrict__ xin, const float* __restrict__ w,
                        float* __restrict__ mean, float* __restrict__ rstd,
                        int Ci, int Co) {
  __shared__ float sh[TPB];
  int co = blockIdx.x;
  float s = 0.f, s2 = 0.f;
  for (int b = 0; b < WB; ++b) {
    const float* xp = xin + (size_t)b * Ci * WN;
    for (int n = threadIdx.x; n < WN; n += TPB) {
      float a = 0.f;
      for (int ci = 0; ci < Ci; ++ci) a += xp[(size_t)ci * WN + n] * w[ci * Co + co];
      s += a; s2 += a * a;
    }
  }
  s  = block_reduce_add(s, sh);
  s2 = block_reduce_add(s2, sh);
  if (threadIdx.x == 0) {
    float inv = 1.f / (float)(WB * WN);
    float m = s * inv;
    float var = s2 * inv - m * m;
    mean[co] = m;
    rstd[co] = rsqrtf(var + 1e-5f);
  }
}

__global__ __launch_bounds__(TPB)
void final_gap_kernel(const float* __restrict__ xin, const float* __restrict__ w,
                      const float* __restrict__ mean, const float* __restrict__ rstd,
                      const float* __restrict__ g, const float* __restrict__ bb,
                      float* __restrict__ out, int Ci, int Co) {
  __shared__ float sh[TPB];
  int co = blockIdx.x % Co;
  int b  = blockIdx.x / Co;
  float m = mean[co], sc = rstd[co] * g[co], bias = bb[co];
  const float* xp = xin + (size_t)b * Ci * WN;
  float s = 0.f;
  for (int n = threadIdx.x; n < WN; n += TPB) {
    float a = 0.f;
    for (int ci = 0; ci < Ci; ++ci) a += xp[(size_t)ci * WN + n] * w[ci * Co + co];
    float v = (a - m) * sc + bias;
    s += fmaxf(v, 0.f);
  }
  s = block_reduce_add(s, sh);
  if (threadIdx.x == 0) out[b * Co + co] = s / (float)WN;
}

// ---------------------------------------------------------------------------
// Host orchestration
// ---------------------------------------------------------------------------
static void run_block(const float* xin, int Ci, int Co, void* const* d_in, int base,
                      float* y, float* xn, float* mean, float* rstd,
                      __bf16* fT, __bf16* gT, __bf16* hbf, float* xout,
                      hipStream_t st) {
  const float* w    = (const float*)d_in[base + 0];
  const float* bias = (const float*)d_in[base + 1];
  const float* bng  = (const float*)d_in[base + 2];
  const float* bnb  = (const float*)d_in[base + 3];
  const float* wfp  = (const float*)d_in[base + 4];
  const float* bfp  = (const float*)d_in[base + 5];
  const float* wgp  = (const float*)d_in[base + 6];
  const float* bgp  = (const float*)d_in[base + 7];
  const float* whp  = (const float*)d_in[base + 8];
  const float* bhp  = (const float*)d_in[base + 9];
  const float* gam  = (const float*)d_in[base + 10];

  int totalY = WB * Co * WN;
  int gY = (totalY + TPB - 1) / TPB;
  conv1x1_kernel<<<gY, TPB, 0, st>>>(xin, w, bias, y, Ci, Co);
  bn_stats_kernel<<<Co, TPB, 0, st>>>(y, mean, rstd, Co);
  bn_relu_kernel<<<gY, TPB, 0, st>>>(y, mean, rstd, bng, bnb, xn, Co);

  int pgrid = WB * WN / TPB;   // 32
  int CF = Co / 8;
  if (CF == 4)
    proj_fg_kernel<4><<<pgrid, TPB, 0, st>>>(xn, wfp, bfp, wgp, bgp, fT, gT, Co);
  else if (CF == 8)
    proj_fg_kernel<8><<<pgrid, TPB, 0, st>>>(xn, wfp, bfp, wgp, bgp, fT, gT, Co);
  else
    proj_fg_kernel<12><<<pgrid, TPB, 0, st>>>(xn, wfp, bfp, wgp, bgp, fT, gT, Co);

  proj_h_kernel<<<gY, TPB, 0, st>>>(xn, whp, bhp, hbf, Co);

  int fgrid = (WB * WN / 16) / 8;   // 512 query tiles / 8 waves = 64 blocks
  int CT = Co / 16;
  if (CT == 2)
    flash_attn_kernel<2><<<fgrid, TPB, 0, st>>>(fT, gT, hbf, xn, gam, xout);
  else if (CT == 4)
    flash_attn_kernel<4><<<fgrid, TPB, 0, st>>>(fT, gT, hbf, xn, gam, xout);
  else
    flash_attn_kernel<6><<<fgrid, TPB, 0, st>>>(fT, gT, hbf, xn, gam, xout);
}

extern "C" void kernel_launch(void* const* d_in, const int* in_sizes, int n_in,
                              void* d_out, int out_size, void* d_ws, size_t ws_size,
                              hipStream_t stream) {
  (void)in_sizes; (void)n_in; (void)out_size; (void)ws_size;
  char* ws = (char*)d_ws;
  size_t off = 0;
  auto take = [&](size_t bytes) -> void* {
    void* p = ws + off;
    off += (bytes + 255) & ~(size_t)255;
    return p;
  };
  float*  y    = (float*)take((size_t)WB * 96 * WN * 4);
  float*  xn   = (float*)take((size_t)WB * 96 * WN * 4);
  float*  pA   = (float*)take((size_t)WB * 96 * WN * 4);
  float*  pB   = (float*)take((size_t)WB * 96 * WN * 4);
  __bf16* fT   = (__bf16*)take((size_t)WB * WN * 32 * 2);
  __bf16* gT   = (__bf16*)take((size_t)WB * WN * 32 * 2);
  __bf16* hbf  = (__bf16*)take((size_t)WB * 96 * WN * 2);
  float*  mean = (float*)take(512 * 4);
  float*  rstd = (float*)take(512 * 4);

  const float* x = (const float*)d_in[0];
  run_block(x,  3,  32, d_in, 1,  y, xn, mean, rstd, fT, gT, hbf, pA, stream);
  run_block(pA, 32, 64, d_in, 12, y, xn, mean, rstd, fT, gT, hbf, pB, stream);
  run_block(pB, 64, 96, d_in, 23, y, xn, mean, rstd, fT, gT, hbf, pA, stream);

  const float* wfin = (const float*)d_in[34];
  final_stats_kernel<<<512, TPB, 0, stream>>>(pA, wfin, mean, rstd, 96, 512);
  final_gap_kernel<<<WB * 512, TPB, 0, stream>>>(pA, wfin, mean, rstd,
      (const float*)d_in[35], (const float*)d_in[36], (float*)d_out, 96, 512);
}
